// GRUModel_63230508532186
// MI455X (gfx1250) — matrix operand order repaired and checked by reading, not compiled, not run on hardware
//
#include <hip/hip_runtime.h>

// ---------------- problem constants ----------------
#define B_TOT   512
#define T_LEN   1000
#define H       64
#define G3      192           // 3*H
#define OUT_N   300
#define BT      64            // batch tile per workgroup
#define NTHREADS 512          // 16 wave32

typedef _Float16 half8 __attribute__((ext_vector_type(8)));
typedef _Float16 v16h  __attribute__((ext_vector_type(16)));
typedef float    v8f   __attribute__((ext_vector_type(8)));

union V16U { v16h v; half8 h[2]; };
union F8U  { float4 v4[2]; float f[8]; };

__device__ __forceinline__ v16h make16(half8 lo, half8 hi){
  V16U u; u.h[0] = lo; u.h[1] = hi; return u.v;
}

__device__ __forceinline__ F8U ldf8(const float* p){   // two 16B loads
  F8U u; u.v4[0] = ((const float4*)p)[0]; u.v4[1] = ((const float4*)p)[1]; return u;
}

// A operand: 16x32 f16 tile from row-major [64][64] f16 LDS buffer.
__device__ __forceinline__ v16h loadA(const _Float16* __restrict__ base, int mt, int k0, int lane){
  int row = mt*16 + (lane & 15);
  int kb  = k0 + ((lane >> 4) << 3);
  const _Float16* p = base + row*H + kb;
  half8 lo = *(const half8*)(p);
  half8 hi = *(const half8*)(p + 16);
  return make16(lo, hi);
}

// B operand: pre-packed so a lane's 16 halves are contiguous (two b128 loads).
__device__ __forceinline__ v16h loadB(const _Float16* __restrict__ mat, int kt, int nt, int lane){
  const _Float16* p = mat + (((kt*12 + nt)*32 + lane) << 4);
  half8 lo = *(const half8*)(p);
  half8 hi = *(const half8*)(p + 8);
  return make16(lo, hi);
}

__device__ __forceinline__ v8f wmma16(v16h a, v16h b, v8f c){
  return __builtin_amdgcn_wmma_f32_16x16x32_f16(false, a, false, b, (short)0, c, false, false);
}

__device__ __forceinline__ float sigf(float x){
  return 1.0f / (1.0f + __expf(-x));
}
__device__ __forceinline__ float tanhfast(float x){
  x = fminf(fmaxf(x, -20.0f), 20.0f);
  float e = __expf(-2.0f * x);
  return (1.0f - e) / (1.0f + e);
}

// -------- prep: pack 5 weight matrices [192][64] f32 -> WMMA-B f16 lane layout --------
__global__ void pack_weights(const float* __restrict__ W0, const float* __restrict__ W1,
                             const float* __restrict__ W2, const float* __restrict__ W3,
                             const float* __restrict__ W4, _Float16* __restrict__ out){
  const float* Ws[5] = {W0, W1, W2, W3, W4};
  const float* W = Ws[blockIdx.x];
  _Float16* o = out + blockIdx.x * 12288;
  for (int c = threadIdx.x; c < 24*32; c += blockDim.x){   // (kt,nt) tile * lane
    int tile = c >> 5, lane = c & 31;
    int kt = tile / 12, nt = tile % 12;
    int n  = nt*16 + (lane & 15);
    int kb = kt*32 + ((lane >> 4) << 3);
    #pragma unroll
    for (int hh = 0; hh < 16; ++hh)
      o[(c << 4) + hh] = (_Float16)W[n*H + kb + hh + ((hh >> 3) << 3)];       // B[k][n] = W[n][k]
  }
}

// -------- persistent 3-layer GRU cell: one WG owns 64 batch rows for all T steps --------
__global__ __launch_bounds__(NTHREADS) void gru_persistent(
    const float* __restrict__ x,        // [512][1000]
    const float* __restrict__ Wih0,     // [192] (IN==1)
    const float* __restrict__ bih0, const float* __restrict__ bhh0,
    const float* __restrict__ bih1, const float* __restrict__ bhh1,
    const float* __restrict__ bih2, const float* __restrict__ bhh2,
    const _Float16* __restrict__ packed,// 5*12288 f16 (Whh0,Wih1,Whh1,Wih2,Whh2)
    float* __restrict__ hlast)          // [512][64]
{
  extern __shared__ char smem[];
  _Float16* sW   = (_Float16*)smem;                // 122880 B : packed weights
  float*    sG   = (float*)(smem + 122880);        //  65536 B : gate scratch
  float*    sH   = (float*)(smem + 188416);        //  49152 B : h f32, 3 layers
  _Float16* sH16 = (_Float16*)(smem + 237568);     //  24576 B : h f16, 3 layers
                                                   //  total 262144 B

  const int tid  = threadIdx.x;
  const int lane = tid & 31, wave = tid >> 5;
  const int mt = wave & 3;        // M tile (batch rows 16*mt..)
  const int ng = wave >> 2;       // N tile group: nt in {3*ng..3*ng+2}
  const int batch0 = blockIdx.x * BT;
  const int cL = (lane & 15);
  const int mb = mt*16 + ((lane >> 4) << 3);

  // ---- init: weights -> LDS, h = 0 ----
  {
    const uint4* src = (const uint4*)packed;
    uint4* dst = (uint4*)sW;
    for (int i = tid; i < 7680; i += NTHREADS) dst[i] = src[i];
    for (int i = tid; i < 3*BT*H; i += NTHREADS){ sH[i] = 0.0f; sH16[i] = (_Float16)0.0f; }
  }

  // ---- loop-invariant register hoists ----
  const int bb  = tid >> 3;        // batch row for elementwise pass
  const int jj0 = (tid & 7) << 3;  // 8 consecutive hidden cols
  // layer-0 input coefficients (xv * W + ...) ; only the n-gate bias must stay separate
  F8U wr  = ldf8(Wih0 + jj0), wz = ldf8(Wih0 + 64 + jj0), wn = ldf8(Wih0 + 128 + jj0);
  F8U b0n = ldf8(bih0 + 128 + jj0);

  // per-lane column biases, added as scalars at store time (WMMA C stays 0)
  float sb0[3], cI1[3], cH1[3], cI2[3], cH2[3];
  #pragma unroll
  for (int i = 0; i < 3; ++i){
    int nt = ng*3 + i;
    int col = nt*16 + cL;
    sb0[i] = bhh0[col] + (nt < 8 ? bih0[col] : 0.0f);  // bih0 foldable for r,z cols only
    cI1[i] = bih1[col]; cH1[i] = bhh1[col];
    cI2[i] = bih2[col]; cH2[i] = bhh2[col];
  }
  __syncthreads();

  const float* xp = x + (size_t)(batch0 + bb) * T_LEN;
  float* sE   = sG;                // [64][128]  r,z pre-activations (incl. all biases)
  float* sGin = sG + 8192;         // [64][64]   gi_n + b_in
  float* sGhn = sG + 12288;        // [64][64]   gh_n + b_hn
  const v8f vzero = {0.f,0.f,0.f,0.f,0.f,0.f,0.f,0.f};

  float xv = xp[0];
  for (int t = 0; t < T_LEN; ++t){
    float xnx = xp[(t + 1 < T_LEN) ? (t + 1) : t];    // prefetch next step's x

    // ======== layer 0 : g = h0 @ Whh0^T (+ biases at store) ========
    {
      v16h a0 = loadA(sH16, mt, 0,  lane);
      v16h a1 = loadA(sH16, mt, 32, lane);
      v16h b0[3], b1[3];
      #pragma unroll
      for (int i = 0; i < 3; ++i){
        int nt = ng*3 + i;
        b0[i] = loadB(sW, 0, nt, lane);
        b1[i] = loadB(sW, 1, nt, lane);
      }
      v8f acc[3];
      #pragma unroll
      for (int i = 0; i < 3; ++i) acc[i] = wmma16(a0, b0[i], vzero);
      #pragma unroll
      for (int i = 0; i < 3; ++i) acc[i] = wmma16(a1, b1[i], acc[i]);
      #pragma unroll
      for (int i = 0; i < 3; ++i){
        int col = (ng*3 + i)*16 + cL;
        float* p = sG + mb*G3 + col;
        #pragma unroll
        for (int r = 0; r < 8; ++r) p[r*G3] = acc[i][r] + sb0[i];
      }
    }
    __syncthreads();
    {
      const float* gb = sG + bb*G3;
      F8U gr = ldf8(gb + jj0), gz = ldf8(gb + 64 + jj0), gn = ldf8(gb + 128 + jj0);
      float* hp = sH + bb*H + jj0;
      F8U h = ldf8(hp);
      F8U hn; half8 h16v;
      #pragma unroll
      for (int q = 0; q < 8; ++q){
        float r = sigf(fmaf(xv, wr.f[q], gr.f[q]));                 // biases already in g
        float z = sigf(fmaf(xv, wz.f[q], gz.f[q]));
        float n = tanhfast(fmaf(xv, wn.f[q], b0n.f[q]) + r*gn.f[q]);
        float hv = (1.0f - z)*n + z*h.f[q];
        hn.f[q] = hv; h16v[q] = (_Float16)hv;
      }
      ((float4*)hp)[0] = hn.v4[0]; ((float4*)hp)[1] = hn.v4[1];
      *(half8*)(sH16 + bb*H + jj0) = h16v;
    }
    __syncthreads();

    // ======== layers 1,2 : gi = inp@Wih^T, gh = h@Whh^T (+ biases at store) ========
    #pragma unroll
    for (int L = 1; L < 3; ++L){
      const _Float16* Wih = sW + (2*L - 1)*12288;
      const _Float16* Whh = sW + (2*L    )*12288;
      const float* cI = (L == 1) ? cI1 : cI2;
      const float* cH = (L == 1) ? cH1 : cH2;
      v8f accI[3], accH[3];
      {
        const _Float16* Ain = sH16 + (L-1)*BT*H;   // prev layer's new h
        v16h a0 = loadA(Ain, mt, 0, lane), a1 = loadA(Ain, mt, 32, lane);
        v16h b0[3], b1[3];
        #pragma unroll
        for (int i = 0; i < 3; ++i){
          int nt = ng*3 + i;
          b0[i] = loadB(Wih, 0, nt, lane);
          b1[i] = loadB(Wih, 1, nt, lane);
        }
        #pragma unroll
        for (int i = 0; i < 3; ++i) accI[i] = wmma16(a0, b0[i], vzero);
        #pragma unroll
        for (int i = 0; i < 3; ++i) accI[i] = wmma16(a1, b1[i], accI[i]);
      }
      {
        const _Float16* Ah = sH16 + L*BT*H;        // this layer's old h
        v16h a0 = loadA(Ah, mt, 0, lane), a1 = loadA(Ah, mt, 32, lane);
        v16h b0[3], b1[3];
        #pragma unroll
        for (int i = 0; i < 3; ++i){
          int nt = ng*3 + i;
          b0[i] = loadB(Whh, 0, nt, lane);
          b1[i] = loadB(Whh, 1, nt, lane);
        }
        #pragma unroll
        for (int i = 0; i < 3; ++i) accH[i] = wmma16(a0, b0[i], vzero);
        #pragma unroll
        for (int i = 0; i < 3; ++i) accH[i] = wmma16(a1, b1[i], accH[i]);
      }
      #pragma unroll
      for (int i = 0; i < 3; ++i){
        int nt = ng*3 + i;
        int col = nt*16 + cL;
        if (nt < 8){                       // r,z gates: gi+gh+biases summed
          float cb = cI[i] + cH[i];
          float* p = sE + mb*128 + col;
          #pragma unroll
          for (int r = 0; r < 8; ++r) p[r*128] = accI[i][r] + accH[i][r] + cb;
        } else {                           // n gate: gi_n / gh_n separate
          float* pg = sGin + mb*H + (col - 128);
          float* ph = sGhn + mb*H + (col - 128);
          #pragma unroll
          for (int r = 0; r < 8; ++r){
            pg[r*H] = accI[i][r] + cI[i];
            ph[r*H] = accH[i][r] + cH[i];
          }
        }
      }
      __syncthreads();
      {
        const float* e = sE + bb*128;
        F8U er = ldf8(e + jj0), ez = ldf8(e + 64 + jj0);
        F8U gi = ldf8(sGin + bb*H + jj0), gh = ldf8(sGhn + bb*H + jj0);
        float* hp = sH + L*BT*H + bb*H + jj0;
        F8U h = ldf8(hp);
        F8U hn; half8 h16v;
        #pragma unroll
        for (int q = 0; q < 8; ++q){
          float r = sigf(er.f[q]);
          float z = sigf(ez.f[q]);
          float n = tanhfast(gi.f[q] + r*gh.f[q]);
          float hv = (1.0f - z)*n + z*h.f[q];
          hn.f[q] = hv; h16v[q] = (_Float16)hv;
        }
        ((float4*)hp)[0] = hn.v4[0]; ((float4*)hp)[1] = hn.v4[1];
        *(half8*)(sH16 + L*BT*H + bb*H + jj0) = h16v;
      }
      __syncthreads();
    }
    xv = xnx;
  }

  // ---- write top-layer hidden state to global workspace ----
  {
    const float* h2 = sH + 2*BT*H;
    for (int i = tid; i < BT*H; i += NTHREADS)
      hlast[(size_t)(batch0 + (i >> 6))*H + (i & 63)] = h2[i];
  }
}

// -------- final FC: out[512][300] = hlast @ fc_w^T + fc_b --------
__global__ void fc_kernel(const float* __restrict__ hlast, const float* __restrict__ fcw,
                          const float* __restrict__ fcb, float* __restrict__ out){
  int idx = blockIdx.x * blockDim.x + threadIdx.x;
  if (idx >= B_TOT * OUT_N) return;
  int b = idx / OUT_N, o = idx - b * OUT_N;
  const float4* hp = (const float4*)(hlast + b*H);
  const float4* wp = (const float4*)(fcw   + o*H);
  float s = fcb[o];
  #pragma unroll
  for (int k = 0; k < H/4; ++k){
    float4 hv = hp[k], wv = wp[k];
    s += hv.x*wv.x + hv.y*wv.y + hv.z*wv.z + hv.w*wv.w;
  }
  out[idx] = s;
}

extern "C" void kernel_launch(void* const* d_in, const int* in_sizes, int n_in,
                              void* d_out, int out_size, void* d_ws, size_t ws_size,
                              hipStream_t stream){
  (void)in_sizes; (void)n_in; (void)out_size; (void)ws_size;
  const float* x    = (const float*)d_in[0];
  const float* Wih0 = (const float*)d_in[1];
  const float* Whh0 = (const float*)d_in[2];
  const float* bih0 = (const float*)d_in[3];
  const float* bhh0 = (const float*)d_in[4];
  const float* Wih1 = (const float*)d_in[5];
  const float* Whh1 = (const float*)d_in[6];
  const float* bih1 = (const float*)d_in[7];
  const float* bhh1 = (const float*)d_in[8];
  const float* Wih2 = (const float*)d_in[9];
  const float* Whh2 = (const float*)d_in[10];
  const float* bih2 = (const float*)d_in[11];
  const float* bhh2 = (const float*)d_in[12];
  const float* fcw  = (const float*)d_in[13];
  const float* fcb  = (const float*)d_in[14];

  char* ws = (char*)d_ws;
  _Float16* packed = (_Float16*)ws;            // 122880 B: packed f16 weights
  float*    hlast  = (float*)(ws + 131072);    // 131072 B: [512][64] f32

  pack_weights<<<5, 256, 0, stream>>>(Whh0, Wih1, Whh1, Wih2, Whh2, packed);

  size_t shmem = 262144;                       // 256 KB < 320 KB WGP LDS
  gru_persistent<<<B_TOT / BT, NTHREADS, shmem, stream>>>(
      x, Wih0, bih0, bhh0, bih1, bhh1, bih2, bhh2, packed, hlast);

  fc_kernel<<<(B_TOT * OUT_N + 255) / 256, 256, 0, stream>>>(hlast, fcw, fcb, (float*)d_out);
}
